// CausalDiffusionModel_1073741824371
// MI455X (gfx1250) — compile-verified, matrix operand order checked by reading, not run on hardware
//
#include <hip/hip_runtime.h>
#include <hip/hip_bf16.h>

#define NB   32
#define DD   1152
#define HEADS 16
#define HD   72
#define DEPTH 12
#define CT   4
#define PP   256
#define PL   16
#define FE   256
#define LL   (CT + PP)          // 260
#define MLPD (4 * DD)           // 4608
#define SCALE 0.1178511301977579f  // 72^-0.5

typedef __attribute__((ext_vector_type(16))) _Float16 v16h;
typedef __attribute__((ext_vector_type(8)))  float    v8f;
typedef __attribute__((ext_vector_type(8)))  unsigned v8u;

__device__ __forceinline__ unsigned pack2h(float a, float b)
{
    unsigned short ua = __builtin_bit_cast(unsigned short, (_Float16)a);
    unsigned short ub = __builtin_bit_cast(unsigned short, (_Float16)b);
    return (unsigned)ua | ((unsigned)ub << 16);
}

// ---------------------------------------------------------------------------
// Tiled GEMM: C[M,N] = act(A[M,K] @ B[K,N] + bias) (+ resid)
// fp32 in, f16 WMMA / f32 accum. 128x64 block, BK=32, 4 waves x (32x64) each.
// LDS tiles stored pre-packed in WMMA fragment order (half2 dwords, [row][K/2])
// so every fragment group is a single aligned ds_load_b128.
// ---------------------------------------------------------------------------
#define BM 128
#define BN 64
#define BK 32
#define KP 16            // BK/2 packed dwords per row
#define LPAD 20          // padded row stride in dwords (80B, 16B aligned)

__global__ __launch_bounds__(128)
void gemm_wmma_f16(const float* __restrict__ A, const float* __restrict__ B,
                   const float* __restrict__ bias, const float* resid,
                   float* C, int M, int N, int K, int act)
{
    __shared__ unsigned AsU[BM][LPAD];   // A tile: [row][kpair]
    __shared__ unsigned BsU[BN][LPAD];   // B tile transposed: [col][kpair]

    const int tid    = threadIdx.x;
    const int wave   = tid >> 5;
    const int lane   = tid & 31;
    const int halfId = lane >> 4;
    const int l16    = lane & 15;
    const int rowBase = blockIdx.y * BM;
    const int colBase = blockIdx.x * BN;

    v8f acc[2][4];
    #pragma unroll
    for (int mr = 0; mr < 2; ++mr)
        #pragma unroll
        for (int nb = 0; nb < 4; ++nb) acc[mr][nb] = (v8f){};

    for (int k0 = 0; k0 < K; k0 += BK) {
        // ---- stage A row (1 row of 32 K per thread), packed to half2 ----
        {
            const int gr = rowBase + tid;
            unsigned ub[KP];
            if (gr < M) {
                const float4* src = (const float4*)(A + (size_t)gr * K + k0);
                #pragma unroll
                for (int q = 0; q < 8; ++q) {
                    float4 f = src[q];
                    ub[2 * q]     = pack2h(f.x, f.y);
                    ub[2 * q + 1] = pack2h(f.z, f.w);
                }
            } else {
                #pragma unroll
                for (int q = 0; q < KP; ++q) ub[q] = 0u;
            }
            uint4* dst = (uint4*)&AsU[tid][0];
            #pragma unroll
            for (int q = 0; q < 4; ++q)
                dst[q] = make_uint4(ub[4 * q], ub[4 * q + 1], ub[4 * q + 2], ub[4 * q + 3]);
        }
        // ---- stage B (2 K-rows x 8 cols per thread), transposed+packed ----
        {
            const int kp = tid >> 3;            // 0..15 -> K rows 2kp, 2kp+1
            const int c0 = (tid & 7) * 8;       // 8 columns
            const float* r0 = B + (size_t)(k0 + 2 * kp) * N + colBase + c0;
            const float* r1 = r0 + N;
            float4 a0 = ((const float4*)r0)[0], a1 = ((const float4*)r0)[1];
            float4 b0 = ((const float4*)r1)[0], b1 = ((const float4*)r1)[1];
            BsU[c0 + 0][kp] = pack2h(a0.x, b0.x);
            BsU[c0 + 1][kp] = pack2h(a0.y, b0.y);
            BsU[c0 + 2][kp] = pack2h(a0.z, b0.z);
            BsU[c0 + 3][kp] = pack2h(a0.w, b0.w);
            BsU[c0 + 4][kp] = pack2h(a1.x, b1.x);
            BsU[c0 + 5][kp] = pack2h(a1.y, b1.y);
            BsU[c0 + 6][kp] = pack2h(a1.z, b1.z);
            BsU[c0 + 7][kp] = pack2h(a1.w, b1.w);
        }
        // ---- prefetch next K tile while waiting ----
        if (k0 + BK < K) {
            const int gr = rowBase + tid;
            if (gr < M) __builtin_prefetch(A + (size_t)gr * K + k0 + BK, 0, 0);
            __builtin_prefetch(B + (size_t)(k0 + BK + 2 * (tid >> 3)) * N +
                               colBase + (tid & 7) * 8, 0, 0);
        }
        __syncthreads();

        // ---- fragments: one ds_load_b128 per group ----
        v16h afr[2];
        #pragma unroll
        for (int mr = 0; mr < 2; ++mr) {
            const int arow = wave * 32 + mr * 16 + l16;
            v8u raw;
            #pragma unroll
            for (int g = 0; g < 2; ++g) {
                uint4 t = *(const uint4*)&AsU[arow][g * 8 + halfId * 4];
                raw[g * 4 + 0] = t.x; raw[g * 4 + 1] = t.y;
                raw[g * 4 + 2] = t.z; raw[g * 4 + 3] = t.w;
            }
            afr[mr] = __builtin_bit_cast(v16h, raw);
        }
        #pragma unroll
        for (int nb = 0; nb < 4; ++nb) {
            const int bcol = nb * 16 + l16;
            v8u raw;
            #pragma unroll
            for (int g = 0; g < 2; ++g) {
                uint4 t = *(const uint4*)&BsU[bcol][g * 8 + halfId * 4];
                raw[g * 4 + 0] = t.x; raw[g * 4 + 1] = t.y;
                raw[g * 4 + 2] = t.z; raw[g * 4 + 3] = t.w;
            }
            v16h bfr = __builtin_bit_cast(v16h, raw);
            #pragma unroll
            for (int mr = 0; mr < 2; ++mr)
                acc[mr][nb] = __builtin_amdgcn_wmma_f32_16x16x32_f16(
                    false, afr[mr], false, bfr, (short)0, acc[mr][nb], false, false);
        }
        __syncthreads();
    }

    // ---- epilogue: bias + activation + residual ----
    #pragma unroll
    for (int nb = 0; nb < 4; ++nb) {
        const int col = colBase + nb * 16 + l16;
        const float bv = bias ? bias[col] : 0.0f;
        #pragma unroll
        for (int mr = 0; mr < 2; ++mr) {
            #pragma unroll
            for (int r = 0; r < 8; ++r) {
                const int grow = rowBase + wave * 32 + mr * 16 + halfId * 8 + r;
                if (grow < M) {
                    float v = acc[mr][nb][r] + bv;
                    if (act == 1) {            // GELU (tanh approx)
                        float x = v;
                        v = 0.5f * x * (1.0f + tanhf(0.7978845608028654f *
                                                     (x + 0.044715f * x * x * x)));
                    } else if (act == 2) {     // SiLU
                        v = v / (1.0f + __expf(-v));
                    }
                    if (resid) v += resid[(size_t)grow * N + col];
                    C[(size_t)grow * N + col] = v;
                }
            }
        }
    }
}

// ---------------------------------------------------------------------------
// Sinusoidal timestep embedding [N, FE]
// ---------------------------------------------------------------------------
__global__ void timestep_embed(const float* __restrict__ t, float* __restrict__ out)
{
    int idx = blockIdx.x * blockDim.x + threadIdx.x;
    if (idx >= NB * FE) return;
    int n = idx / FE, i = idx % FE;
    const int hf = FE / 2;
    int j = (i < hf) ? i : i - hf;
    float freq = __expf(-9.210340371976184f * (float)j / (float)hf); // ln(1e4)
    float a = t[n] * freq;
    out[idx] = (i < hf) ? cosf(a) : sinf(a);
}

// ---------------------------------------------------------------------------
// Conditioning tokens: tok[n, c, :] = ytab[y[n]][c] + cpos[c] + temb[n]
// ---------------------------------------------------------------------------
__global__ void cond_tokens(const int* __restrict__ y, const float* __restrict__ ytab,
                            const float* __restrict__ cpos, const float* __restrict__ temb,
                            float* __restrict__ tok)
{
    int idx = blockIdx.x * blockDim.x + threadIdx.x;
    if (idx >= NB * CT * DD) return;
    int d = idx % DD;
    int c = (idx / DD) % CT;
    int n = idx / (CT * DD);
    int lab = y[n];
    tok[((size_t)n * LL + c) * DD + d] =
        ytab[(size_t)lab * (DD * CT) + c * DD + d] + cpos[c * DD + d] + temb[(size_t)n * DD + d];
}

// ---------------------------------------------------------------------------
// Patch embedding: tok[n, 4+p, d] = x[n,p,:] @ xw[:,d] + xb[d] + pos[p,d]
// ---------------------------------------------------------------------------
__global__ void patch_embed(const float* __restrict__ x, const float* __restrict__ xw,
                            const float* __restrict__ xb, const float* __restrict__ pos,
                            float* __restrict__ tok)
{
    int idx = blockIdx.x * blockDim.x + threadIdx.x;
    if (idx >= NB * PP * DD) return;
    int d = idx % DD;
    int p = (idx / DD) % PP;
    int n = idx / (PP * DD);
    const float* xr = x + ((size_t)n * PP + p) * PL;
    float acc = xb[d] + pos[(size_t)p * DD + d];
    #pragma unroll
    for (int j = 0; j < PL; ++j) acc += xr[j] * xw[j * DD + d];
    tok[((size_t)n * LL + (CT + p)) * DD + d] = acc;
}

// ---------------------------------------------------------------------------
// LayerNorm over D=1152, one block per token row
// ---------------------------------------------------------------------------
__global__ __launch_bounds__(256)
void layernorm(const float* __restrict__ in, const float* __restrict__ g,
               const float* __restrict__ b, float* __restrict__ out)
{
    __shared__ float r1[256];
    __shared__ float r2[256];
    const int row = blockIdx.x;
    const float* xr = in + (size_t)row * DD;
    float s = 0.0f, s2 = 0.0f;
    for (int d = threadIdx.x; d < DD; d += 256) {
        float v = xr[d]; s += v; s2 += v * v;
    }
    r1[threadIdx.x] = s; r2[threadIdx.x] = s2;
    __syncthreads();
    for (int off = 128; off > 0; off >>= 1) {
        if (threadIdx.x < off) { r1[threadIdx.x] += r1[threadIdx.x + off];
                                 r2[threadIdx.x] += r2[threadIdx.x + off]; }
        __syncthreads();
    }
    float mean = r1[0] * (1.0f / DD);
    float var  = r2[0] * (1.0f / DD) - mean * mean;
    float rstd = rsqrtf(var + 1e-5f);
    float* orow = out + (size_t)row * DD;
    for (int d = threadIdx.x; d < DD; d += 256)
        orow[d] = (xr[d] - mean) * rstd * g[d] + b[d];
}

// ---------------------------------------------------------------------------
// Per-head q/k layernorm (in-place on qkv); q additionally scaled by HD^-0.5
// ---------------------------------------------------------------------------
__global__ void qknorm(float* qkv, const float* __restrict__ qg, const float* __restrict__ qb,
                       const float* __restrict__ kg, const float* __restrict__ kb)
{
    int idx = blockIdx.x * blockDim.x + threadIdx.x;
    if (idx >= NB * LL * HEADS) return;
    int h   = idx % HEADS;
    int row = idx / HEADS;                 // n*L + l
    float* q = qkv + (size_t)row * (3 * DD) + h * HD;
    float* k = q + DD;
    {
        float s = 0.0f;
        for (int d = 0; d < HD; ++d) s += q[d];
        float m = s * (1.0f / HD);
        float v = 0.0f;
        for (int d = 0; d < HD; ++d) { float t = q[d] - m; v += t * t; }
        float r = rsqrtf(v * (1.0f / HD) + 1e-5f);
        for (int d = 0; d < HD; ++d)
            q[d] = ((q[d] - m) * r * qg[d] + qb[d]) * SCALE;
    }
    {
        float s = 0.0f;
        for (int d = 0; d < HD; ++d) s += k[d];
        float m = s * (1.0f / HD);
        float v = 0.0f;
        for (int d = 0; d < HD; ++d) { float t = k[d] - m; v += t * t; }
        float r = rsqrtf(v * (1.0f / HD) + 1e-5f);
        for (int d = 0; d < HD; ++d)
            k[d] = (k[d] - m) * r * kg[d] + kb[d];
    }
}

// ---------------------------------------------------------------------------
// Causal attention, online softmax. One block per (n, head); thread = query row.
// K/V staged in LDS in chunks of 64 rows.
// ---------------------------------------------------------------------------
#define ATT_BLK 288
#define KCH 64

__global__ __launch_bounds__(ATT_BLK)
void attention(const float* __restrict__ qkv, float* __restrict__ obuf)
{
    __shared__ float ks[KCH][HD];
    __shared__ float vs[KCH][HD];
    const int n = blockIdx.x / HEADS;
    const int h = blockIdx.x % HEADS;
    const int t = threadIdx.x;
    const size_t rs = 3 * DD;
    const float* base = qkv + (size_t)n * LL * rs;

    float q[HD], o[HD];
    float m = -1e30f, ssum = 0.0f;
    const int i = t;
    const bool active = (i < LL);
    if (active) {
        const float* qr = base + (size_t)i * rs + h * HD;
        for (int d = 0; d < HD; ++d) { q[d] = qr[d]; o[d] = 0.0f; }
    }

    for (int j0 = 0; j0 < LL; j0 += KCH) {
        __syncthreads();
        for (int e = t; e < KCH * HD; e += ATT_BLK) {
            int jj = e / HD, d = e % HD;
            int gj = j0 + jj;
            if (gj < LL) {
                const float* kr = base + (size_t)gj * rs + DD + h * HD;
                ks[jj][d] = kr[d];
                vs[jj][d] = kr[DD + d];
            }
        }
        __syncthreads();
        if (active && j0 <= i) {
            int jmax = i - j0 + 1;
            if (jmax > KCH) jmax = KCH;
            for (int jj = 0; jj < jmax; ++jj) {
                float a = 0.0f;
                for (int d = 0; d < HD; ++d) a += q[d] * ks[jj][d];
                if (a > m) {
                    float corr = __expf(m - a);
                    ssum = ssum * corr + 1.0f;
                    for (int d = 0; d < HD; ++d) o[d] = o[d] * corr + vs[jj][d];
                    m = a;
                } else {
                    float p = __expf(a - m);
                    ssum += p;
                    for (int d = 0; d < HD; ++d) o[d] += p * vs[jj][d];
                }
            }
        }
    }

    if (active) {
        float inv = 1.0f / ssum;
        float* orow = obuf + ((size_t)n * LL + i) * DD + h * HD;
        for (int d = 0; d < HD; ++d) orow[d] = o[d] * inv;
    }
}

// ---------------------------------------------------------------------------
// Output head: out[n,p,:] = h[n, 4+p, :] @ ow + ob   (N=16 cols, scalar kernel)
// ---------------------------------------------------------------------------
__global__ void head_proj(const float* __restrict__ hb, const float* __restrict__ ow,
                          const float* __restrict__ ob, float* __restrict__ out)
{
    int idx = blockIdx.x * blockDim.x + threadIdx.x;
    if (idx >= NB * PP * PL) return;
    int j = idx % PL;
    int p = (idx / PL) % PP;
    int n = idx / (PP * PL);
    const float* hr = hb + ((size_t)n * LL + CT + p) * DD;
    float acc = ob[j];
    for (int d = 0; d < DD; ++d) acc += hr[d] * ow[d * PL + j];
    out[idx] = acc;
}

// ---------------------------------------------------------------------------
extern "C" void kernel_launch(void* const* d_in, const int* in_sizes, int n_in,
                              void* d_out, int out_size, void* d_ws, size_t ws_size,
                              hipStream_t stream)
{
    const float* x    = (const float*)d_in[0];
    const float* t    = (const float*)d_in[1];
    const int*   y    = (const int*)  d_in[2];
    const float* xw   = (const float*)d_in[3];
    const float* xb   = (const float*)d_in[4];
    const float* tw1  = (const float*)d_in[5];
    const float* tb1  = (const float*)d_in[6];
    const float* tw2  = (const float*)d_in[7];
    const float* tb2  = (const float*)d_in[8];
    const float* ytab = (const float*)d_in[9];
    const float* pos  = (const float*)d_in[10];
    const float* cpos = (const float*)d_in[11];
    const float* ln1g = (const float*)d_in[12];
    const float* ln1b = (const float*)d_in[13];
    const float* qkvw = (const float*)d_in[14];
    const float* qng  = (const float*)d_in[15];
    const float* qnb  = (const float*)d_in[16];
    const float* kng  = (const float*)d_in[17];
    const float* knb  = (const float*)d_in[18];
    const float* pw   = (const float*)d_in[19];
    const float* pb   = (const float*)d_in[20];
    const float* ln2g = (const float*)d_in[21];
    const float* ln2b = (const float*)d_in[22];
    const float* f1w  = (const float*)d_in[23];
    const float* f1b  = (const float*)d_in[24];
    const float* f2w  = (const float*)d_in[25];
    const float* f2b  = (const float*)d_in[26];
    const float* fng  = (const float*)d_in[27];
    const float* fnb  = (const float*)d_in[28];
    const float* ow   = (const float*)d_in[29];
    const float* ob   = (const float*)d_in[30];

    const size_t tokN = (size_t)NB * LL * DD;        // 9,584,640 floats
    float* tok   = (float*)d_ws;
    float* hbuf  = tok + tokN;
    float* R     = hbuf + tokN;                      // shared region, 4*tokN floats
    float* qkv   = R;                                // [N*L, 3D]
    float* obuf  = R + (size_t)NB * LL * 3 * DD;     // [N*L, D]
    float* mlp   = R;                                // [N*L, MLPD] (aliases qkv+obuf)
    float* temb0 = R + 4 * tokN;                     // [N, FE]
    float* t1    = temb0 + (size_t)NB * FE;          // [N, D]
    float* temb  = t1 + (size_t)NB * DD;             // [N, D]

    const int rowsTok = NB * LL;                     // 8320

    // --- timestep embedding + MLP ---
    timestep_embed<<<(NB * FE + 255) / 256, 256, 0, stream>>>(t, temb0);
    gemm_wmma_f16<<<dim3(DD / BN, (NB + BM - 1) / BM), 128, 0, stream>>>(
        temb0, tw1, tb1, nullptr, t1, NB, DD, FE, 2 /*silu*/);
    gemm_wmma_f16<<<dim3(DD / BN, (NB + BM - 1) / BM), 128, 0, stream>>>(
        t1, tw2, tb2, nullptr, temb, NB, DD, DD, 0);

    // --- token assembly ---
    cond_tokens<<<(NB * CT * DD + 255) / 256, 256, 0, stream>>>(y, ytab, cpos, temb, tok);
    patch_embed<<<(NB * PP * DD + 255) / 256, 256, 0, stream>>>(x, xw, xb, pos, tok);

    // --- transformer blocks ---
    for (int l = 0; l < DEPTH; ++l) {
        layernorm<<<rowsTok, 256, 0, stream>>>(tok, ln1g + l * DD, ln1b + l * DD, hbuf);
        gemm_wmma_f16<<<dim3(3 * DD / BN, rowsTok / BM), 128, 0, stream>>>(
            hbuf, qkvw + (size_t)l * DD * 3 * DD, nullptr, nullptr,
            qkv, rowsTok, 3 * DD, DD, 0);
        qknorm<<<(NB * LL * HEADS + 255) / 256, 256, 0, stream>>>(
            qkv, qng + l * HD, qnb + l * HD, kng + l * HD, knb + l * HD);
        attention<<<NB * HEADS, ATT_BLK, 0, stream>>>(qkv, obuf);
        gemm_wmma_f16<<<dim3(DD / BN, rowsTok / BM), 128, 0, stream>>>(
            obuf, pw + (size_t)l * DD * DD, pb + l * DD, tok,
            tok, rowsTok, DD, DD, 0);
        layernorm<<<rowsTok, 256, 0, stream>>>(tok, ln2g + l * DD, ln2b + l * DD, hbuf);
        gemm_wmma_f16<<<dim3(MLPD / BN, rowsTok / BM), 128, 0, stream>>>(
            hbuf, f1w + (size_t)l * DD * MLPD, f1b + l * MLPD, nullptr,
            mlp, rowsTok, MLPD, DD, 1 /*gelu*/);
        gemm_wmma_f16<<<dim3(DD / BN, rowsTok / BM), 128, 0, stream>>>(
            mlp, f2w + (size_t)l * MLPD * DD, f2b + l * DD, tok,
            tok, rowsTok, DD, MLPD, 0);
    }

    // --- final norm + head ---
    layernorm<<<rowsTok, 256, 0, stream>>>(tok, fng, fnb, hbuf);
    head_proj<<<(NB * PP * PL + 255) / 256, 256, 0, stream>>>(hbuf, ow, ob, (float*)d_out);
}
